// SevenSacredNeurons_81922206204364
// MI455X (gfx1250) — matrix operand order repaired and checked by reading, not compiled, not run on hardware
//
#include <hip/hip_runtime.h>
#include <hip/hip_bf16.h>

typedef __attribute__((ext_vector_type(16))) __bf16 v16bf;
typedef __attribute__((ext_vector_type(8)))  __bf16 v8bf;
typedef __attribute__((ext_vector_type(8)))  float  v8f;
typedef __attribute__((ext_vector_type(4)))  float  v4f;

// phi_spiral(v) = sigmoid(v/PHI) * (v*PHI / (1+|v|))
//              = v*PHI * rcp( (1 + exp(-v/PHI)) * (1 + |v|) )
// Single v_rcp_f32 + v_exp_f32 instead of two IEEE division expansions:
// the activation VALU cost is co-limiting with HBM here (~31M evals).
__device__ __forceinline__ float phi_spiral(float v) {
  const float PHI     = 1.6180339887498949f;
  const float INV_PHI = 0.6180339887498949f;
  float t     = __expf(-v * INV_PHI);                 // v_mul + v_exp_f32
  float denom = (1.0f + t) * (1.0f + fabsf(v));       // |v| folds into src modifier
  return (v * PHI) * __builtin_amdgcn_rcpf(denom);    // v_rcp_f32 (~1 ULP)
}

// Fused SevenSacredNeurons forward.
// One 256-thread block = 8 waves; each wave owns a 16-row M-tile.
// All weights live in LDS as zero-padded bf16 B-operands ([N][K] layout so a
// B-fragment is one contiguous 32B LDS load). Activations round-trip through
// a per-wave LDS slice between GEMMs (same-wave DS ops are in-order on CDNA5).
__global__ __launch_bounds__(256) void seven_sacred_wmma(
    const float* __restrict__ x,
    const float* __restrict__ W_in, const float* __restrict__ b_in,
    const float* __restrict__ Wn,   const float* __restrict__ bn,
    const float* __restrict__ W_int,const float* __restrict__ b_int,
    const float* __restrict__ gamma,const float* __restrict__ beta,
    const float* __restrict__ W_out,const float* __restrict__ b_out,
    float* __restrict__ out, int numTiles, int B)
{
  __shared__ __attribute__((aligned(32))) __bf16 sWin[64 * 256];  // GEMM1 B: [N=64][K=256]
  __shared__ __attribute__((aligned(32))) __bf16 sW2 [64 * 64];   // GEMM2 B: [N=64][K=64]
  __shared__ __attribute__((aligned(32))) __bf16 sW3 [32 * 64];   // GEMM3 B: [N=32][K=64]
  __shared__ __attribute__((aligned(16))) float  sSmall[256];     // biases / gamma / beta / W_out / b_out
  __shared__ __attribute__((aligned(32))) __bf16 sScr[8 * 16 * 64]; // per-wave activation staging (2KB/wave)

  const int tid = threadIdx.x;

  // ---- one-time weight preload (zero-padded) ----
  for (int i = tid; i < 64 * 256; i += 256) {
    int c = i >> 8, k = i & 255;
    sWin[i] = (c < 49) ? (__bf16)W_in[c * 256 + k] : (__bf16)0.0f;
  }
  for (int i = tid; i < 64 * 64; i += 256) {
    int j = i >> 6, k = i & 63;
    sW2[i] = (j < 49 && k < 49) ? (__bf16)Wn[j * 49 + k] : (__bf16)0.0f;
  }
  for (int i = tid; i < 32 * 64; i += 256) {
    int j = i >> 6, k = i & 63;
    sW3[i] = (j < 21 && k < 49) ? (__bf16)W_int[j * 49 + k] : (__bf16)0.0f;
  }
  if (tid < 49) sSmall[tid]       = b_in[tid];
  if (tid < 49) sSmall[49 + tid]  = bn[tid];     // bn flattened [7,7] -> [49]
  if (tid < 21) sSmall[98 + tid]  = b_int[tid];
  if (tid < 21) sSmall[119 + tid] = gamma[tid];
  if (tid < 21) sSmall[140 + tid] = beta[tid];
  if (tid < 84) sSmall[161 + tid] = W_out[tid];  // [4,21] row-major
  if (tid < 4)  sSmall[245 + tid] = b_out[tid];
  __syncthreads();

  const int wave = tid >> 5;
  const int lane = tid & 31;
  const int m16  = lane & 15;
  const int hi   = lane >> 4;

  __bf16* hl = &sScr[wave * 1024];   // 16 rows x 64 cols bf16 (this wave's slice)
  float*  il = (float*)hl;           // reused: 16 rows x 24 floats for LayerNorm stage

  for (int tile = blockIdx.x; tile < numTiles; tile += gridDim.x) {
    const int rowBase = tile * 128 + wave * 16;
    int row = rowBase + m16;
    if (row >= B) row = B - 1;
    const float* xrow = x + (size_t)row * 256;

    // prefetch next tile's x rows into cache hierarchy
    {
      int nt = tile + (int)gridDim.x;
      if (nt < numTiles) {
        int prow = nt * 128 + wave * 16 + m16;
        if (prow < B) __builtin_prefetch(x + (size_t)prow * 256, 0, 1);
      }
    }

    // ================= GEMM1: [16,256] x [256,64] (N real = 49) =================
    v8f acc0 = {}, acc1 = {}, acc2 = {}, acc3 = {};
#pragma unroll
    for (int kt = 0; kt < 8; kt++) {
      // A-fragment: lane<16 holds K = b..b+15 (b = kt*32); lane>=16 shifted by 16
      const int kbA = kt * 32 + hi * 16;
      v4f f0 = *(const v4f*)(xrow + kbA);
      v4f f1 = *(const v4f*)(xrow + kbA + 4);
      v4f f2 = *(const v4f*)(xrow + kbA + 8);
      v4f f3 = *(const v4f*)(xrow + kbA + 12);
      v16bf a;
#pragma unroll
      for (int e = 0; e < 4; e++) {
        a[e]      = (__bf16)f0[e];
        a[4 + e]  = (__bf16)f1[e];
        a[8 + e]  = (__bf16)f2[e];
        a[12 + e] = (__bf16)f3[e];
      }
      // B-fragment: lane = column, half selects K half; contiguous 32B in LDS
      const int kbB = kt * 32 + hi * 16;
      v16bf b0 = *(const v16bf*)&sWin[(0 * 16 + m16) * 256 + kbB];
      v16bf b1 = *(const v16bf*)&sWin[(1 * 16 + m16) * 256 + kbB];
      v16bf b2 = *(const v16bf*)&sWin[(2 * 16 + m16) * 256 + kbB];
      v16bf b3 = *(const v16bf*)&sWin[(3 * 16 + m16) * 256 + kbB];
      acc0 = __builtin_amdgcn_wmma_f32_16x16x32_bf16(false, a, false, b0, (short)0, acc0, false, false);
      acc1 = __builtin_amdgcn_wmma_f32_16x16x32_bf16(false, a, false, b1, (short)0, acc1, false, false);
      acc2 = __builtin_amdgcn_wmma_f32_16x16x32_bf16(false, a, false, b2, (short)0, acc2, false, false);
      acc3 = __builtin_amdgcn_wmma_f32_16x16x32_bf16(false, a, false, b3, (short)0, acc3, false, false);
    }

    // bias + phi_spiral, stage h[16,64] (zero pad cols >= 49) to this wave's LDS slice
    auto store_act = [&](const v8f& acc, int nt, int biasOfs) {
      int col = nt * 16 + m16;
      bool valid = col < 49;
      float bias = valid ? sSmall[biasOfs + col] : 0.0f;
#pragma unroll
      for (int r = 0; r < 8; r++) {
        int rm = r + hi * 8;
        float s = phi_spiral(acc[r] + bias);
        hl[rm * 64 + col] = valid ? (__bf16)s : (__bf16)0.0f;
      }
    };
    store_act(acc0, 0, 0); store_act(acc1, 1, 0);
    store_act(acc2, 2, 0); store_act(acc3, 3, 0);

    // ================= GEMM2: [16,64] x [64,64] (K,N real = 49) =================
    v8f z0 = {}, z1 = {}, z2 = {}, z3 = {};
#pragma unroll
    for (int kt = 0; kt < 2; kt++) {
      const int kbA = kt * 32 + hi * 16;
      v8bf a0 = *(const v8bf*)&hl[m16 * 64 + kbA];
      v8bf a1 = *(const v8bf*)&hl[m16 * 64 + kbA + 8];
      v16bf a;
#pragma unroll
      for (int e = 0; e < 8; e++) { a[e] = a0[e]; a[8 + e] = a1[e]; }
      const int kbB = kt * 32 + hi * 16;
      v16bf b0 = *(const v16bf*)&sW2[(0 * 16 + m16) * 64 + kbB];
      v16bf b1 = *(const v16bf*)&sW2[(1 * 16 + m16) * 64 + kbB];
      v16bf b2 = *(const v16bf*)&sW2[(2 * 16 + m16) * 64 + kbB];
      v16bf b3 = *(const v16bf*)&sW2[(3 * 16 + m16) * 64 + kbB];
      z0 = __builtin_amdgcn_wmma_f32_16x16x32_bf16(false, a, false, b0, (short)0, z0, false, false);
      z1 = __builtin_amdgcn_wmma_f32_16x16x32_bf16(false, a, false, b1, (short)0, z1, false, false);
      z2 = __builtin_amdgcn_wmma_f32_16x16x32_bf16(false, a, false, b2, (short)0, z2, false, false);
      z3 = __builtin_amdgcn_wmma_f32_16x16x32_bf16(false, a, false, b3, (short)0, z3, false, false);
    }
    // bias(bn) + phi_spiral -> combined[16,64] back into same slice (same-wave DS in order)
    store_act(z0, 0, 49); store_act(z1, 1, 49);
    store_act(z2, 2, 49); store_act(z3, 3, 49);

    // ================= GEMM3: [16,64] x [64,32] (N real = 21) =================
    v8f g0 = {}, g1 = {};
#pragma unroll
    for (int kt = 0; kt < 2; kt++) {
      const int kbA = kt * 32 + hi * 16;
      v8bf a0 = *(const v8bf*)&hl[m16 * 64 + kbA];
      v8bf a1 = *(const v8bf*)&hl[m16 * 64 + kbA + 8];
      v16bf a;
#pragma unroll
      for (int e = 0; e < 8; e++) { a[e] = a0[e]; a[8 + e] = a1[e]; }
      const int kbB = kt * 32 + hi * 16;
      v16bf b0 = *(const v16bf*)&sW3[(0 * 16 + m16) * 64 + kbB];
      v16bf b1 = *(const v16bf*)&sW3[(1 * 16 + m16) * 64 + kbB];
      g0 = __builtin_amdgcn_wmma_f32_16x16x32_bf16(false, a, false, b0, (short)0, g0, false, false);
      g1 = __builtin_amdgcn_wmma_f32_16x16x32_bf16(false, a, false, b1, (short)0, g1, false, false);
    }

    // integrated = phi_spiral(. + b_int): store 21 floats/row (stride 24 to soften bank conflicts)
    {
      int col0 = m16;            // g0 columns 0..15
      float b0f = sSmall[98 + col0];
#pragma unroll
      for (int r = 0; r < 8; r++)
        il[(r + hi * 8) * 24 + col0] = phi_spiral(g0[r] + b0f);
      int col1 = 16 + m16;       // g1 columns 16..31, only 16..20 valid
      if (col1 < 21) {
        float b1f = sSmall[98 + col1];
#pragma unroll
        for (int r = 0; r < 8; r++)
          il[(r + hi * 8) * 24 + col1] = phi_spiral(g1[r] + b1f);
      }
    }

    // ============ LayerNorm(21) + [21]x[21,4] output: one row per lane 0..15 ============
    if (lane < 16) {
      const int r = lane;
      const int orow = rowBase + r;
      if (orow < B) {
        float v[21];
        float mean = 0.0f;
#pragma unroll
        for (int j = 0; j < 21; j++) { v[j] = il[r * 24 + j]; mean += v[j]; }
        mean *= (1.0f / 21.0f);
        float var = 0.0f;
#pragma unroll
        for (int j = 0; j < 21; j++) { float d = v[j] - mean; var += d * d; }
        var *= (1.0f / 21.0f);
        float inv = rsqrtf(var + 1e-5f);
        float o0 = sSmall[245], o1 = sSmall[246], o2 = sSmall[247], o3 = sSmall[248];
#pragma unroll
        for (int j = 0; j < 21; j++) {
          float n = (v[j] - mean) * inv * sSmall[119 + j] + sSmall[140 + j];
          o0 = fmaf(n, sSmall[161 + 0 * 21 + j], o0);
          o1 = fmaf(n, sSmall[161 + 1 * 21 + j], o1);
          o2 = fmaf(n, sSmall[161 + 2 * 21 + j], o2);
          o3 = fmaf(n, sSmall[161 + 3 * 21 + j], o3);
        }
        v4f res = {o0, o1, o2, o3};
        *(v4f*)(out + (size_t)orow * 4) = res;
      }
    }
  }
}

extern "C" void kernel_launch(void* const* d_in, const int* in_sizes, int n_in,
                              void* d_out, int out_size, void* d_ws, size_t ws_size,
                              hipStream_t stream) {
  (void)n_in; (void)out_size; (void)d_ws; (void)ws_size;
  const float* x     = (const float*)d_in[0];
  const float* W_in  = (const float*)d_in[1];
  const float* b_in  = (const float*)d_in[2];
  const float* Wn    = (const float*)d_in[3];
  const float* bn    = (const float*)d_in[4];
  const float* W_int = (const float*)d_in[5];
  const float* b_int = (const float*)d_in[6];
  const float* gamma = (const float*)d_in[7];
  const float* beta  = (const float*)d_in[8];
  const float* W_out = (const float*)d_in[9];
  const float* b_out = (const float*)d_in[10];
  float* out = (float*)d_out;

  const int B = in_sizes[0] / 256;                // 262144
  const int numTiles = (B + 127) / 128;           // 128 rows per block-tile
  int grid = numTiles < 512 ? numTiles : 512;     // grid-stride over tiles

  hipLaunchKernelGGL(seven_sacred_wmma, dim3(grid), dim3(256), 0, stream,
                     x, W_in, b_in, Wn, bn, W_int, b_int, gamma, beta,
                     W_out, b_out, out, numTiles, B);
}
